// LiTrALayerMem_11166914969786
// MI455X (gfx1250) — matrix operand order repaired and checked by reading, not compiled, not run on hardware
//
#include <hip/hip_runtime.h>

typedef __bf16 bf16;
typedef __attribute__((ext_vector_type(16))) __bf16 v16bf;
typedef __attribute__((ext_vector_type(8)))  __bf16 v8bf;
typedef __attribute__((ext_vector_type(8)))  float   v8f;
typedef __attribute__((ext_vector_type(4)))  unsigned int u32x4;
typedef __attribute__((ext_vector_type(8)))  int     i32x8;
typedef __attribute__((ext_vector_type(4)))  int     i32x4;

#define NTOK   8192
#define DMODEL 256
#define SCALE_Q 0.17677669529663687f   // 32^-0.5

#define HAVE_TDM (__has_builtin(__builtin_amdgcn_tensor_load_to_lds) && \
                  __has_builtin(__builtin_amdgcn_s_wait_tensorcnt))

// ---------------------------------------------------------------- WMMA core
__device__ __forceinline__ v8f wmma_bf16(v16bf a, v16bf b, v8f c) {
  return __builtin_amdgcn_wmma_f32_16x16x32_bf16(false, a, false, b, (short)0, c,
                                                 false, false);
}

// A-operand fragment (16x32 bf16) from a row-major row pointer.
// lane<16: K = {k0..k0+7, k0+16..k0+23}; lane>=16: shifted by 8 (hi = lane>>4)
__device__ __forceinline__ v16bf ldA(const bf16* rowp, int k0, int hi) {
  const bf16* p = rowp + k0 + (hi << 3);
  v8bf lo = *(const v8bf*)(p);
  v8bf hb = *(const v8bf*)(p + 16);
  v16bf r;
#pragma unroll
  for (int e = 0; e < 8; ++e) { r[e] = lo[e]; r[e + 8] = hb[e]; }
  return r;
}

// B-operand fragment (32x16 bf16) from row-major [N][K]: same K order as ldA
__device__ __forceinline__ v16bf ldB(const bf16* rowp, int k0, int hi) {
  const bf16* p = rowp + k0 + (hi << 3);
  v8bf lo = *(const v8bf*)(p);
  v8bf hb = *(const v8bf*)(p + 16);
  v16bf r;
#pragma unroll
  for (int e = 0; e < 8; ++e) { r[e] = lo[e]; r[e + 8] = hb[e]; }
  return r;
}

__device__ __forceinline__ float red16_max(float v) {
#pragma unroll
  for (int m = 1; m <= 8; m <<= 1) v = fmaxf(v, __shfl_xor(v, m, 32));
  return v;
}
__device__ __forceinline__ float red16_sum(float v) {
#pragma unroll
  for (int m = 1; m <= 8; m <<= 1) v += __shfl_xor(v, m, 32);
  return v;
}

// token index for plane regrouping: plane 0 = HW (g=frame), 1 = HT, 2 = WT
__device__ __forceinline__ int plane_tok(int g, int n, int plane) {
  if (plane == 0) return (g << 8) + n;
  int b = g >> 4, gl = g & 15, t = n >> 4, nl = n & 15;
  return plane == 1 ? (b << 12) + (t << 8) + (nl << 4) + gl
                    : (b << 12) + (t << 8) + (gl << 4) + nl;
}

#if HAVE_TDM
// Tensor Data Mover: DMA a row-major [rows][rowlen] bf16 tile (contiguous rows)
// from global memory into LDS at byte offset lds_off. D# layout per CDNA5 ISA
// CDNA5 ISA (2D tensor, groups 2/3 zero = unused).
// This toolchain exposes the 6-arg builtin form (extra int32x8, zero-filled).
__device__ __forceinline__ void tdm_load_2d(unsigned lds_off, const void* gptr,
                                            unsigned rows, unsigned rowlen) {
  unsigned long long ga = (unsigned long long)(uintptr_t)gptr;
  u32x4 g0;
  g0[0] = 1u;                                             // count=1, user mode
  g0[1] = lds_off;                                        // lds_addr (bytes)
  g0[2] = (unsigned)(ga & 0xffffffffu);                   // global_addr lo
  g0[3] = ((unsigned)(ga >> 32) & 0x01ffffffu) | (2u << 30); // addr hi | type=2
  i32x8 g1;
  g1[0] = (int)(1u << 16);                                // data_size=2B; no mask
  g1[1] = (int)((rowlen & 0xffffu) << 16);                // tensor_dim0[15:0]
  g1[2] = (int)(((rowlen >> 16) & 0xffffu) | ((rows & 0xffffu) << 16)); // d0 hi|d1 lo
  g1[3] = (int)(((rows >> 16) & 0xffffu) | ((rowlen & 0xffffu) << 16)); // d1 hi|tile_dim0
  g1[4] = (int)(rows & 0xffffu);                          // tile_dim1 (tile_dim2=0)
  g1[5] = (int)rowlen;                                    // tensor_dim0_stride lo
  g1[6] = 0;                                              // stride hi | d1_stride lo
  g1[7] = 0;                                              // d1_stride hi
  i32x4 z4 = {0, 0, 0, 0};
  i32x8 z8 = {0, 0, 0, 0, 0, 0, 0, 0};
  __builtin_amdgcn_tensor_load_to_lds(g0, g1, z4, z4, z8, 0);
}
#endif

// ---------------------------------------------------------------- GEMM
// C[M,N] = act(A_bf16[M,K] @ W + bias) * alpha (+ residual); W given as WT[N][K] bf16
// B (weight) panel is DMA'd into LDS via the Tensor Data Mover; A streams from
// global, interleaving global_load_b128 with v_wmma.
__global__ __launch_bounds__(32) void k_gemm(
    const bf16* __restrict__ A, const bf16* __restrict__ WT,
    const float* __restrict__ bias, int bias_n,
    float* __restrict__ Cf, bf16* __restrict__ Cbf,
    const float* __restrict__ res,
    int M, int N, int K, float alpha, int act) {
  extern __shared__ char smem_raw[];
  bf16* lb = (bf16*)smem_raw;                 // [16][K] weight panel
  int lane = threadIdx.x;
  int l15 = lane & 15, hi = lane >> 4;
  int n0 = blockIdx.x << 4, m0 = blockIdx.y << 4;
  const bf16* arow = A + (size_t)(m0 + l15) * K;
  v8f acc = {};
#if HAVE_TDM
  tdm_load_2d(0u /* panel at LDS base */, WT + (size_t)n0 * K, 16u, (unsigned)K);
  __builtin_amdgcn_s_wait_tensorcnt((short)0);
  asm volatile("" ::: "memory");              // keep ds reads after the wait
  const bf16* brow = lb + (size_t)l15 * K;
#else
  const bf16* brow = WT + (size_t)(n0 + l15) * K;
#endif
  for (int k = 0; k < K; k += 32)
    acc = wmma_bf16(ldA(arow, k, hi), ldB(brow, k, hi), acc);
  int col = n0 + l15;
  float bcol = (bias && col < bias_n) ? bias[col] : 0.0f;
#pragma unroll
  for (int r = 0; r < 8; ++r) {
    float v = acc[r] + bcol;
    if (act == 1) v = v / (1.0f + expf(-v));                                  // silu
    else if (act == 2) v = 0.5f * v * (1.0f + erff(v * 0.70710678118654752f)); // gelu
    v *= alpha;
    size_t off = (size_t)(m0 + (hi << 3) + r) * (size_t)N + col;
    if (res) v += res[off];
    if (Cf)  Cf[off] = v;
    if (Cbf) Cbf[off] = (bf16)v;
  }
}

// ---------------------------------------------------------------- LayerNorm (D=256)
__global__ __launch_bounds__(256) void k_ln(
    const float* __restrict__ x, const float* __restrict__ g,
    const float* __restrict__ b, bf16* __restrict__ out, int rows) {
  int row = (blockIdx.x << 3) + (threadIdx.x >> 5);
  int lane = threadIdx.x & 31;
  if (row >= rows) return;
  const float* xr = x + (size_t)row * DMODEL;
  float v[8]; float s = 0.0f;
#pragma unroll
  for (int i = 0; i < 8; ++i) { v[i] = xr[(i << 5) + lane]; s += v[i]; }
#pragma unroll
  for (int m = 1; m <= 16; m <<= 1) s += __shfl_xor(s, m, 32);
  float mu = s * (1.0f / 256.0f);
  float q = 0.0f;
#pragma unroll
  for (int i = 0; i < 8; ++i) { float d = v[i] - mu; q += d * d; }
#pragma unroll
  for (int m = 1; m <= 16; m <<= 1) q += __shfl_xor(q, m, 32);
  float rs = rsqrtf(q * (1.0f / 256.0f) + 1e-5f);
  bf16* orow = out + (size_t)row * DMODEL;
#pragma unroll
  for (int i = 0; i < 8; ++i) {
    int c = (i << 5) + lane;
    orow[c] = (bf16)((v[i] - mu) * rs * g[c] + b[c]);
  }
}

// ---------------------------------------------------------------- small prep kernels
__global__ void k_transpose(const float* __restrict__ W, bf16* __restrict__ WT,
                            int K, int N, int total) {
  int idx = blockIdx.x * 256 + threadIdx.x;
  if (idx >= total) return;
  int n = idx / K, kk = idx - n * K;
  WT[idx] = (n < N) ? (bf16)W[(size_t)kk * N + n] : (bf16)0.0f;
}

__global__ void k_tobf16(const float* __restrict__ in, bf16* __restrict__ out, int total) {
  int idx = blockIdx.x * 256 + threadIdx.x;
  if (idx < total) out[idx] = (bf16)in[idx];
}

__global__ void k_mqin(const float* __restrict__ mem, const float* __restrict__ memt,
                       bf16* __restrict__ out, int m, int total) {
  int idx = blockIdx.x * 256 + threadIdx.x;
  if (idx >= total) return;
  int d = idx & 255;
  int r = (idx >> 8) % m;
  int t = idx / (m << 8);
  out[idx] = (bf16)(mem[(r << 8) + d] + memt[(t << 8) + d]);
}

// VT[((g*8+h)*32+dh)*256 + n] = Vproj[tok(g,n)][h*32+dh]   (per-plane gathered V^T)
__global__ void k_vt(const bf16* __restrict__ v, bf16* __restrict__ vt, int plane) {
  int idx = blockIdx.x * 256 + threadIdx.x;           // < 2^21
  int n = idx & 255, dh = (idx >> 8) & 31, h = (idx >> 13) & 7, g = idx >> 16;
  vt[idx] = v[(size_t)plane_tok(g, n, plane) * DMODEL + h * 32 + dh];
}

__global__ void k_fuse(const float* __restrict__ gate, const float* __restrict__ mhw,
                       const float* __restrict__ mht, const float* __restrict__ mwt,
                       bf16* __restrict__ fused) {
  int idx = blockIdx.x * 256 + threadIdx.x;           // token*8 + h
  int token = idx >> 3, h = idx & 7;
  const float* gr = gate + (size_t)token * 32 + h * 3;
  float a0 = gr[0], a1 = gr[1], a2 = gr[2];
  float mx = fmaxf(a0, fmaxf(a1, a2));
  float e0 = __expf(a0 - mx), e1 = __expf(a1 - mx), e2 = __expf(a2 - mx);
  float inv = 1.0f / (e0 + e1 + e2);
  e0 *= inv; e1 *= inv; e2 *= inv;
  size_t base = (size_t)token * DMODEL + h * 32;
#pragma unroll
  for (int dh = 0; dh < 32; ++dh) {
    size_t o = base + dh;
    fused[o] = (bf16)(e0 * mhw[o] + e1 * mht[o] + e2 * mwt[o]);
  }
}

// ---------------------------------------------------------------- HW plane attention
// per block: (frame bt, head h). m=128 memory slots, N=256 tokens, DH=32.
__global__ __launch_bounds__(256) void k_attn_hw(
    const bf16* __restrict__ q, const bf16* __restrict__ k,
    const bf16* __restrict__ mq, const bf16* __restrict__ mk,
    const bf16* __restrict__ vt, float* __restrict__ out) {
  __shared__ __align__(32) bf16 Pb[128 * 256];     // Bp row-major [m][n]   64KB
  __shared__ __align__(32) bf16 BVT[32 * 128];     // (Bp.V)^T [dh][m]       8KB
  __shared__ __align__(32) bf16 Aw[8][16 * 128];   // per-wave A strip      32KB
  int bt = blockIdx.x >> 3, h = blockIdx.x & 7;
  int lane = threadIdx.x & 31, wave = threadIdx.x >> 5;
  int l15 = lane & 15, hi = lane >> 4;

  // Phase 1: Bp = softmax_n( (mq*scale) . k^T ), one 16-row strip per wave
  {
    int m0 = wave << 4;
    v16bf a = ldA(mq + (size_t)(m0 + l15) * DMODEL + h * 32, 0, hi);
    v8f s[16];
#pragma unroll
    for (int nt = 0; nt < 16; ++nt) {
      v16bf bb = ldB(k + (size_t)((bt << 8) + (nt << 4) + l15) * DMODEL + h * 32, 0, hi);
      v8f z = {};
      s[nt] = wmma_bf16(a, bb, z);
    }
    float mx[8], sm[8];
#pragma unroll
    for (int r = 0; r < 8; ++r) mx[r] = -1e30f;
#pragma unroll
    for (int nt = 0; nt < 16; ++nt)
#pragma unroll
      for (int r = 0; r < 8; ++r) mx[r] = fmaxf(mx[r], s[nt][r]);
#pragma unroll
    for (int r = 0; r < 8; ++r) { mx[r] = red16_max(mx[r]); sm[r] = 0.0f; }
#pragma unroll
    for (int nt = 0; nt < 16; ++nt)
#pragma unroll
      for (int r = 0; r < 8; ++r) { float e = __expf(s[nt][r] - mx[r]); s[nt][r] = e; sm[r] += e; }
#pragma unroll
    for (int r = 0; r < 8; ++r) sm[r] = 1.0f / red16_sum(sm[r]);
#pragma unroll
    for (int nt = 0; nt < 16; ++nt)
#pragma unroll
      for (int r = 0; r < 8; ++r)
        Pb[(size_t)(m0 + (hi << 3) + r) * 256 + (nt << 4) + l15] = (bf16)(s[nt][r] * sm[r]);
  }
  __syncthreads();

  // Phase 2: BV = Bp . V  -> store transposed bf16
  {
    int m0 = wave << 4;
#pragma unroll
    for (int nt = 0; nt < 2; ++nt) {
      v8f acc = {};
#pragma unroll
      for (int kc = 0; kc < 256; kc += 32) {
        v16bf a = ldA(&Pb[(size_t)(m0 + l15) * 256], kc, hi);
        v16bf b = ldB(vt + (size_t)(((bt << 3) + h) * 32 + (nt << 4) + l15) * 256, kc, hi);
        acc = wmma_bf16(a, b, acc);
      }
      int dh = (nt << 4) + l15;
#pragma unroll
      for (int r = 0; r < 8; ++r)
        BVT[dh * 128 + m0 + (hi << 3) + r] = (bf16)acc[r];
    }
  }
  __syncthreads();

  // Phase 3: out = softmax_m((q*scale).mk^T) . BV, two 16-row strips per wave
  for (int it = 0; it < 2; ++it) {
    int row0 = ((it << 3) + wave) << 4;
    v16bf a = ldA(q + (size_t)((bt << 8) + row0 + l15) * DMODEL + h * 32, 0, hi);
    v8f s[8];
#pragma unroll
    for (int mt = 0; mt < 8; ++mt) {
      v16bf b = ldB(mk + (size_t)((mt << 4) + l15) * DMODEL + h * 32, 0, hi);
      v8f z = {};
      s[mt] = wmma_bf16(a, b, z);
    }
    float mx[8], sm[8];
#pragma unroll
    for (int r = 0; r < 8; ++r) mx[r] = -1e30f;
#pragma unroll
    for (int mt = 0; mt < 8; ++mt)
#pragma unroll
      for (int r = 0; r < 8; ++r) mx[r] = fmaxf(mx[r], s[mt][r]);
#pragma unroll
    for (int r = 0; r < 8; ++r) { mx[r] = red16_max(mx[r]); sm[r] = 0.0f; }
#pragma unroll
    for (int mt = 0; mt < 8; ++mt)
#pragma unroll
      for (int r = 0; r < 8; ++r) { float e = __expf(s[mt][r] - mx[r]); s[mt][r] = e; sm[r] += e; }
#pragma unroll
    for (int r = 0; r < 8; ++r) sm[r] = 1.0f / red16_sum(sm[r]);
#pragma unroll
    for (int mt = 0; mt < 8; ++mt)
#pragma unroll
      for (int r = 0; r < 8; ++r)
        Aw[wave][((hi << 3) + r) * 128 + (mt << 4) + l15] = (bf16)(s[mt][r] * sm[r]);
    __builtin_amdgcn_wave_barrier();   // same-wave LDS ordering (DS in-order)
#pragma unroll
    for (int nt = 0; nt < 2; ++nt) {
      v8f acc = {};
#pragma unroll
      for (int kc = 0; kc < 128; kc += 32) {
        v16bf aa = ldA(&Aw[wave][l15 * 128], kc, hi);
        v16bf bb = ldB(&BVT[((nt << 4) + l15) * 128], kc, hi);
        acc = wmma_bf16(aa, bb, acc);
      }
      int col = h * 32 + (nt << 4) + l15;
#pragma unroll
      for (int r = 0; r < 8; ++r)
        out[(size_t)((bt << 8) + row0 + (hi << 3) + r) * DMODEL + col] = acc[r];
    }
    __builtin_amdgcn_wave_barrier();
  }
}

// ---------------------------------------------------------------- causal planes
// per block: (group g, head h). m=64 memory slots, T=16, blk=16, N=256.
__global__ __launch_bounds__(256) void k_attn_causal(
    const bf16* __restrict__ q, const bf16* __restrict__ k,
    const bf16* __restrict__ mq, const bf16* __restrict__ mk,
    const bf16* __restrict__ vt, float* __restrict__ out, int plane) {
  __shared__ __align__(32) bf16 Ab[256 * 64];      // A [n][m]        32KB
  __shared__ __align__(32) bf16 Pb[64 * 256];      // P_t [m][n]      32KB
  __shared__ __align__(32) bf16 ST[16 * 32 * 64];  // S_t^T [t][dh][m] 64KB
  int g = blockIdx.x >> 3, h = blockIdx.x & 7;
  int lane = threadIdx.x & 31, wave = threadIdx.x >> 5;
  int l15 = lane & 15, hi = lane >> 4;

  // Phase 1: A = softmax_m( (q*scale) . mk^T ) -> Ab[n][m]
  for (int it = 0; it < 2; ++it) {
    int n0 = ((it << 3) + wave) << 4;
    v16bf a = ldA(q + (size_t)plane_tok(g, n0 + l15, plane) * DMODEL + h * 32, 0, hi);
    v8f s[4];
#pragma unroll
    for (int mt = 0; mt < 4; ++mt) {
      v16bf b = ldB(mk + (size_t)((mt << 4) + l15) * DMODEL + h * 32, 0, hi);
      v8f z = {};
      s[mt] = wmma_bf16(a, b, z);
    }
    float mx[8], sm[8];
#pragma unroll
    for (int r = 0; r < 8; ++r) mx[r] = -1e30f;
#pragma unroll
    for (int mt = 0; mt < 4; ++mt)
#pragma unroll
      for (int r = 0; r < 8; ++r) mx[r] = fmaxf(mx[r], s[mt][r]);
#pragma unroll
    for (int r = 0; r < 8; ++r) { mx[r] = red16_max(mx[r]); sm[r] = 0.0f; }
#pragma unroll
    for (int mt = 0; mt < 4; ++mt)
#pragma unroll
      for (int r = 0; r < 8; ++r) { float e = __expf(s[mt][r] - mx[r]); s[mt][r] = e; sm[r] += e; }
#pragma unroll
    for (int r = 0; r < 8; ++r) sm[r] = 1.0f / red16_sum(sm[r]);
#pragma unroll
    for (int mt = 0; mt < 4; ++mt)
#pragma unroll
      for (int r = 0; r < 8; ++r)
        Ab[(size_t)(n0 + (hi << 3) + r) * 64 + (mt << 4) + l15] = (bf16)(s[mt][r] * sm[r]);
  }
  __syncthreads();

  // Phase 2: per time-step t: P_t (causal-masked softmax), then S_t = P_t . V
  for (int t = 0; t < 16; ++t) {
    if (wave < 4) {               // wave-uniform branch: EXEC stays all-ones
      int m0 = wave << 4;
      v16bf a = ldA(mq + (size_t)((t << 6) + m0 + l15) * DMODEL + h * 32, 0, hi);
      v8f s[16];
#pragma unroll
      for (int nt = 0; nt < 16; ++nt) {
        if (nt <= t) {
          v16bf b = ldB(k + (size_t)plane_tok(g, (nt << 4) + l15, plane) * DMODEL + h * 32, 0, hi);
          v8f z = {};
          s[nt] = wmma_bf16(a, b, z);
        } else {
#pragma unroll
          for (int r = 0; r < 8; ++r) s[nt][r] = -1e30f;   // causal mask (tile granular)
        }
      }
      float mx[8], sm[8];
#pragma unroll
      for (int r = 0; r < 8; ++r) mx[r] = -1e30f;
#pragma unroll
      for (int nt = 0; nt < 16; ++nt)
#pragma unroll
        for (int r = 0; r < 8; ++r) mx[r] = fmaxf(mx[r], s[nt][r]);
#pragma unroll
      for (int r = 0; r < 8; ++r) { mx[r] = red16_max(mx[r]); sm[r] = 0.0f; }
#pragma unroll
      for (int nt = 0; nt < 16; ++nt)
#pragma unroll
        for (int r = 0; r < 8; ++r) { float e = __expf(s[nt][r] - mx[r]); s[nt][r] = e; sm[r] += e; }
#pragma unroll
      for (int r = 0; r < 8; ++r) sm[r] = 1.0f / red16_sum(sm[r]);
#pragma unroll
      for (int nt = 0; nt < 16; ++nt)   // masked tiles store exact zeros -> padded K ok
#pragma unroll
        for (int r = 0; r < 8; ++r)
          Pb[(size_t)(m0 + (hi << 3) + r) * 256 + (nt << 4) + l15] = (bf16)(s[nt][r] * sm[r]);
    }
    __syncthreads();
    {
      int m0 = (wave >> 1) << 4;
      int dt = (wave & 1) << 4;
      int kmax = (((t + 1) << 4) + 31) & ~31;
      v8f acc = {};
      for (int kc = 0; kc < kmax; kc += 32) {
        v16bf a = ldA(&Pb[(size_t)(m0 + l15) * 256], kc, hi);
        v16bf b = ldB(vt + (size_t)(((g << 3) + h) * 32 + dt + l15) * 256, kc, hi);
        acc = wmma_bf16(a, b, acc);
      }
      int dh = dt + l15;
#pragma unroll
      for (int r = 0; r < 8; ++r)
        ST[((t << 5) + dh) * 64 + m0 + (hi << 3) + r] = (bf16)acc[r];
    }
    __syncthreads();
  }

  // Phase 3: out[t-block] = A rows . S_t
  for (int task = wave; task < 32; task += 8) {
    int t = task >> 1, dt = (task & 1) << 4;
    v8f acc = {};
#pragma unroll
    for (int kc = 0; kc < 64; kc += 32) {
      v16bf a = ldA(&Ab[(size_t)((t << 4) + l15) * 64], kc, hi);
      v16bf b = ldB(&ST[((t << 5) + dt + l15) * 64], kc, hi);
      acc = wmma_bf16(a, b, acc);
    }
    int col = h * 32 + dt + l15;
#pragma unroll
    for (int r = 0; r < 8; ++r)
      out[(size_t)plane_tok(g, (t << 4) + (hi << 3) + r, plane) * DMODEL + col] = acc[r];
  }
}

// ---------------------------------------------------------------- host orchestration
extern "C" void kernel_launch(void* const* d_in, const int* in_sizes, int n_in,
                              void* d_out, int out_size, void* d_ws, size_t ws_size,
                              hipStream_t stream) {
  (void)in_sizes; (void)n_in; (void)out_size; (void)ws_size;
  auto F = [&](int i) { return (const float*)d_in[i]; };
  const float* x = F(0);
  const float* qw[3] = { F(1),  F(7),  F(13) };
  const float* qb[3] = { F(2),  F(8),  F(14) };
  const float* kw[3] = { F(3),  F(9),  F(15) };
  const float* kb[3] = { F(4),  F(10), F(16) };
  const float* vw[3] = { F(5),  F(11), F(17) };
  const float* vb[3] = { F(6),  F(12), F(18) };
  const float* mem_hw = F(19); const float* mem_ht = F(20); const float* mem_wt = F(21);
  const float* memt_ht = F(22); const float* memt_wt = F(23);
  const float* g1_w = F(24); const float* g1_b = F(25);
  const float* g2_w = F(26); const float* g2_b = F(27);
  const float* out_w = F(28); const float* out_b = F(29);
  const float* ln1_g = F(30); const float* ln1_b = F(31);
  const float* ln2_g = F(32); const float* ln2_b = F(33);
  const float* fc1_w = F(34); const float* fc1_b = F(35);
  const float* fc2_w = F(36); const float* fc2_b = F(37);

  char* wsp = (char*)d_ws;
  auto alloc = [&](size_t bytes) -> void* {
    void* r = (void*)wsp; wsp += (bytes + 255) & ~(size_t)255; return r;
  };
  const size_t NB = (size_t)NTOK * DMODEL;

  bf16* ybf = (bf16*)alloc(NB * 2);
  bf16* WTq[3], *WTk[3], *WTv[3];
  for (int p = 0; p < 3; ++p) {
    WTq[p] = (bf16*)alloc(256 * 256 * 2);
    WTk[p] = (bf16*)alloc(256 * 256 * 2);
    WTv[p] = (bf16*)alloc(256 * 256 * 2);
  }
  bf16* WTg1  = (bf16*)alloc(256 * 256 * 2);
  bf16* WTg2  = (bf16*)alloc(32 * 256 * 2);
  bf16* WTout = (bf16*)alloc(256 * 256 * 2);
  bf16* WTfc1 = (bf16*)alloc(1024 * 256 * 2);
  bf16* WTfc2 = (bf16*)alloc(256 * 1024 * 2);
  bf16 *qp[3], *kp[3], *vp[3], *vtb[3]; float* mf[3];
  for (int p = 0; p < 3; ++p) {
    qp[p]  = (bf16*)alloc(NB * 2);
    kp[p]  = (bf16*)alloc(NB * 2);
    vp[p]  = (bf16*)alloc(NB * 2);
    vtb[p] = (bf16*)alloc(NB * 2);
    mf[p]  = (float*)alloc(NB * 4);
  }
  bf16* g1o = (bf16*)alloc(NB * 2);
  float* g2o = (float*)alloc((size_t)NTOK * 32 * 4);
  bf16* memhw_bf = (bf16*)alloc(128 * 256 * 2);
  bf16* memht_bf = (bf16*)alloc(64 * 256 * 2);
  bf16* memwt_bf = (bf16*)alloc(64 * 256 * 2);
  bf16* mqhw = (bf16*)alloc(128 * 256 * 2);
  bf16* mkhw = (bf16*)alloc(128 * 256 * 2);
  bf16* mkht = (bf16*)alloc(64 * 256 * 2);
  bf16* mkwt = (bf16*)alloc(64 * 256 * 2);
  bf16* mqin_ht = (bf16*)alloc(1024 * 256 * 2);
  bf16* mqp_ht  = (bf16*)alloc(1024 * 256 * 2);
  bf16* mqin_wt = (bf16*)alloc(1024 * 256 * 2);
  bf16* mqp_wt  = (bf16*)alloc(1024 * 256 * 2);
  bf16* fusedb = (bf16*)alloc(NB * 2);
  float* x1 = (float*)alloc(NB * 4);
  bf16* hbf = (bf16*)alloc(NB * 2);
  bf16* h1bf = (bf16*)alloc((size_t)NTOK * 1024 * 2);

  auto transpose = [&](const float* W, bf16* WT, int K, int N, int Npad) {
    int total = Npad * K;
    k_transpose<<<(total + 255) / 256, 256, 0, stream>>>(W, WT, K, N, total);
  };
  auto gemm = [&](const bf16* A, const bf16* WT, const float* bias, int bias_n,
                  float* Cf, bf16* Cbf, const float* res,
                  int M, int N, int K, float alpha, int act) {
    // dynamic LDS = one 16xK bf16 weight panel for the TDM path
    k_gemm<<<dim3(N / 16, M / 16), 32, (unsigned)(16 * K * 2), stream>>>(
        A, WT, bias, bias_n, Cf, Cbf, res, M, N, K, alpha, act);
  };

  // ---- LN1 + weight prep
  k_ln<<<NTOK / 8, 256, 0, stream>>>(x, ln1_g, ln1_b, ybf, NTOK);
  for (int p = 0; p < 3; ++p) {
    transpose(qw[p], WTq[p], 256, 256, 256);
    transpose(kw[p], WTk[p], 256, 256, 256);
    transpose(vw[p], WTv[p], 256, 256, 256);
  }
  transpose(g1_w, WTg1, 256, 256, 256);
  transpose(g2_w, WTg2, 256, 24, 32);
  transpose(out_w, WTout, 256, 256, 256);
  transpose(fc1_w, WTfc1, 256, 1024, 1024);
  transpose(fc2_w, WTfc2, 1024, 256, 256);
  k_tobf16<<<128, 256, 0, stream>>>(mem_hw, memhw_bf, 128 * 256);
  k_tobf16<<<64, 256, 0, stream>>>(mem_ht, memht_bf, 64 * 256);
  k_tobf16<<<64, 256, 0, stream>>>(mem_wt, memwt_bf, 64 * 256);
  k_mqin<<<1024, 256, 0, stream>>>(mem_ht, memt_ht, mqin_ht, 64, 16 * 64 * 256);
  k_mqin<<<1024, 256, 0, stream>>>(mem_wt, memt_wt, mqin_wt, 64, 16 * 64 * 256);

  // ---- QKV projections (q pre-scaled by 1/sqrt(DH))
  for (int p = 0; p < 3; ++p) {
    gemm(ybf, WTq[p], qb[p], 256, nullptr, qp[p], nullptr, NTOK, 256, 256, SCALE_Q, 0);
    gemm(ybf, WTk[p], kb[p], 256, nullptr, kp[p], nullptr, NTOK, 256, 256, 1.0f, 0);
    gemm(ybf, WTv[p], vb[p], 256, nullptr, vp[p], nullptr, NTOK, 256, 256, 1.0f, 0);
  }
  // ---- memory-side projections
  gemm(memhw_bf, WTq[0], qb[0], 256, nullptr, mqhw, nullptr, 128, 256, 256, SCALE_Q, 0);
  gemm(memhw_bf, WTk[0], kb[0], 256, nullptr, mkhw, nullptr, 128, 256, 256, 1.0f, 0);
  gemm(memht_bf, WTk[1], kb[1], 256, nullptr, mkht, nullptr, 64, 256, 256, 1.0f, 0);
  gemm(memwt_bf, WTk[2], kb[2], 256, nullptr, mkwt, nullptr, 64, 256, 256, 1.0f, 0);
  gemm(mqin_ht, WTq[1], qb[1], 256, nullptr, mqp_ht, nullptr, 1024, 256, 256, SCALE_Q, 0);
  gemm(mqin_wt, WTq[2], qb[2], 256, nullptr, mqp_wt, nullptr, 1024, 256, 256, SCALE_Q, 0);
  // ---- gate MLP
  gemm(ybf, WTg1, g1_b, 256, nullptr, g1o, nullptr, NTOK, 256, 256, 1.0f, 1);
  gemm(g1o, WTg2, g2_b, 24, g2o, nullptr, nullptr, NTOK, 32, 256, 1.0f, 0);
  // ---- V^T gathers per plane
  for (int p = 0; p < 3; ++p)
    k_vt<<<NTOK, 256, 0, stream>>>(vp[p], vtb[p], p);
  // ---- attention
  k_attn_hw<<<256, 256, 0, stream>>>(qp[0], kp[0], mqhw, mkhw, vtb[0], mf[0]);
  k_attn_causal<<<256, 256, 0, stream>>>(qp[1], kp[1], mqp_ht, mkht, vtb[1], mf[1], 1);
  k_attn_causal<<<256, 256, 0, stream>>>(qp[2], kp[2], mqp_wt, mkwt, vtb[2], mf[2], 2);
  // ---- fuse + out-proj + residual
  k_fuse<<<NTOK * 8 / 256, 256, 0, stream>>>(g2o, mf[0], mf[1], mf[2], fusedb);
  gemm(fusedb, WTout, out_b, 256, x1, nullptr, x, NTOK, 256, 256, 1.0f, 0);
  // ---- LN2 + MLP + residual -> d_out
  k_ln<<<NTOK / 8, 256, 0, stream>>>(x1, ln2_g, ln2_b, hbf, NTOK);
  gemm(hbf, WTfc1, fc1_b, 1024, nullptr, h1bf, nullptr, NTOK, 1024, 256, 1.0f, 2);
  gemm(h1bf, WTfc2, fc2_b, 256, (float*)d_out, nullptr, x1, NTOK, 256, 1024, 1.0f, 0);
}
